// EGNNDecoderLayer_5832565588032
// MI455X (gfx1250) — matrix operand order repaired and checked by reading, not compiled, not run on hardware
//
#include <hip/hip_runtime.h>

// ---------------------------------------------------------------------------
// EGNN decoder layer for MI455X (gfx1250, wave32, WMMA bf16->f32).
// All GEMMs use V_WMMA_F32_16X16X32_BF16. Messages are never materialized in
// HBM: edge MLP -> atomic segment-sum -> coord MLP fused per 16-edge tile.
// Round 2: f32->bf16 conversion via hardware cvt (vector fptrunc) instead of
// manual bit-twiddling, cutting the VALU tax in the gather/store stages.
// ---------------------------------------------------------------------------

typedef unsigned short u16;
typedef __attribute__((ext_vector_type(16))) __bf16 v16bf;
typedef __attribute__((ext_vector_type(2)))  __bf16 v2bf;
typedef __attribute__((ext_vector_type(8)))  float  v8f;
typedef __attribute__((ext_vector_type(2)))  float  v2f;

#define D    128          // feature dim (fixed by the model)
#define WPB  4            // waves per block
#define TPB  (WPB * 32)

// packed f32x2 -> bf16x2 (lowers to v_cvt_pk_bf16_f32 when available)
__device__ __forceinline__ unsigned cvt2(float x, float y) {
  v2f f = {x, y};
  v2bf b = __builtin_convertvector(f, v2bf);
  union { v2bf v; unsigned u; } c; c.v = b;
  return c.u;
}

__device__ __forceinline__ u16 cvt1(float x) {
  union { __bf16 b; u16 u; } c; c.b = (__bf16)x;
  return c.u;
}

// A-operand fragment (16x32 bf16). Per ISA layout: lanes 0-15 hold row M=lane,
// K = 0..7 and 16..23; lanes 16-31 hold K = 8..15 and 24..31.
// p points at the first 8 elements; the second 8 live 16 elements later.
__device__ __forceinline__ v16bf frag_a(const u16* p) {
  union { v16bf v; uint4 q[2]; } f;
  f.q[0] = *(const uint4*)(p);
  f.q[1] = *(const uint4*)(p + 16);
  return f.v;
}

// B-operand fragment (32x16 bf16). Lanes 0-15: col N=lane, K=0..7 and 16..23;
// lanes 16-31: col N=lane-16, K=8..15 and 24..31 (same K order as frag_a).
// p points at the first 8 elements; the second 8 live 16 elements later.
__device__ __forceinline__ v16bf frag_b(const u16* p) {
  union { v16bf v; uint4 q[2]; } f;
  f.q[0] = *(const uint4*)(p);
  f.q[1] = *(const uint4*)(p + 16);
  return f.v;
}

__device__ __forceinline__ v8f wmma_bf16(v16bf a, v16bf b, v8f c) {
  return __builtin_amdgcn_wmma_f32_16x16x32_bf16(false, a, false, b,
                                                 (short)0, c, false, false);
}

// ---------------------------------------------------------------------------
// Weight prep: 7 128x128 blocks -> transposed [out][in] bf16 in workspace.
//   m=0: We1[0:128]   m=1: We1[128:256]  m=2: We2  m=3: Wc1
//   m=4: Wn1[0:128]   m=5: Wn1[128:256]  m=6: Wn2
// ---------------------------------------------------------------------------
__global__ void prep_weights(const float* __restrict__ We1,
                             const float* __restrict__ We2,
                             const float* __restrict__ Wc1,
                             const float* __restrict__ Wn1,
                             const float* __restrict__ Wn2,
                             u16* __restrict__ Wt) {
  int i = blockIdx.x * blockDim.x + threadIdx.x;
  if (i >= 7 * D * D) return;
  int m = i / (D * D);
  int r = i % (D * D);
  int n = r / D, k = r % D;
  float v;
  switch (m) {
    case 0: v = We1[(size_t)k * D + n];        break;
    case 1: v = We1[(size_t)(D + k) * D + n];  break;
    case 2: v = We2[(size_t)k * D + n];        break;
    case 3: v = Wc1[(size_t)k * D + n];        break;
    case 4: v = Wn1[(size_t)k * D + n];        break;
    case 5: v = Wn1[(size_t)(D + k) * D + n];  break;
    default: v = Wn2[(size_t)k * D + n];       break;
  }
  Wt[(size_t)m * D * D + (size_t)n * D + k] = cvt1(v);
}

// Zero agg scratch; seed output coords with input coords (d_out is poisoned).
__global__ void init_out(const float* __restrict__ coords,
                         float* __restrict__ agg,
                         float* __restrict__ out_c, int N) {
  int i = blockIdx.x * blockDim.x + threadIdx.x;
  if (i < N * D) agg[i] = 0.0f;
  if (i < N * 3) out_c[i] = coords[i];
}

// ---------------------------------------------------------------------------
// Edge kernel: one wave32 per 16-edge tile (E is a multiple of 16).
// ---------------------------------------------------------------------------
__global__ __launch_bounds__(TPB) void edge_kernel(
    const float* __restrict__ h, const float* __restrict__ coords,
    const int* __restrict__ eidx,
    const float* __restrict__ We1,   // row 256 = dist2 weights
    const float* __restrict__ be1, const float* __restrict__ be2,
    const float* __restrict__ bc1, const float* __restrict__ Wc2,
    const u16* __restrict__ Wt,
    float* __restrict__ agg, float* __restrict__ out_c,
    int Ne, int nTiles) {
  __shared__ u16  Ar [WPB][16 * D];     // h[row] tile, bf16
  __shared__ u16  Ac [WPB][16 * D];     // h[col] tile, bf16
  __shared__ u16  Hid[WPB][16 * D];     // relu(layer1) tile, bf16
  __shared__ int  Rw [WPB][16];
  __shared__ int  Cl [WPB][16];
  __shared__ float Rel[WPB][16][3];
  __shared__ float D2 [WPB][16];

  const int w  = threadIdx.x >> 5;
  const int L  = threadIdx.x & 31;
  const int lo = L & 15;
  const int hi = L >> 4;
  const int tile = blockIdx.x * WPB + w;
  if (tile >= nTiles) return;

  u16* Msg = Ar[w];   // alias: Ar is dead after layer 1

  // --- stage edge metadata (one edge per lane 0..15) ---
  if (L < 16) {
    int e = tile * 16 + L;
    if (e >= Ne) e = Ne - 1;            // E%16==0 in practice
    int r = eidx[e];
    int c = eidx[Ne + e];
    Rw[w][L] = r; Cl[w][L] = c;
    float dx = coords[(size_t)r * 3 + 0] - coords[(size_t)c * 3 + 0];
    float dy = coords[(size_t)r * 3 + 1] - coords[(size_t)c * 3 + 1];
    float dz = coords[(size_t)r * 3 + 2] - coords[(size_t)c * 3 + 2];
    Rel[w][L][0] = dx; Rel[w][L][1] = dy; Rel[w][L][2] = dz;
    D2[w][L] = dx * dx + dy * dy + dz * dz;
  }

  // --- gather h[row], h[col] into LDS as bf16 (4 floats per lane per row) ---
  for (int r = 0; r < 16; ++r) {
    int er = Rw[w][r];
    int ec = Cl[w][r];
    float4 xr = *(const float4*)(h + (size_t)er * D + 4 * L);
    float4 xc = *(const float4*)(h + (size_t)ec * D + 4 * L);
    uint2 pr, pc;
    pr.x = cvt2(xr.x, xr.y);
    pr.y = cvt2(xr.z, xr.w);
    pc.x = cvt2(xc.x, xc.y);
    pc.y = cvt2(xc.z, xc.w);
    *(uint2*)(&Ar[w][r * D + 4 * L]) = pr;
    *(uint2*)(&Ac[w][r * D + 4 * L]) = pc;
  }

  // per-lane copies for the 8 output rows this lane owns (M = g + 8*hi)
  float d2v[8]; int rg[8];
#pragma unroll
  for (int g = 0; g < 8; ++g) {
    d2v[g] = D2[w][8 * hi + g];
    rg[g]  = Rw[w][8 * hi + g];
  }
  const float* w1d = We1 + (size_t)(2 * D) * D;   // We1 row 256 (dist2)

  // --- layer 1: Hid = relu(h_r@W1a + h_c@W1b + dist2 (x) w1d + be1) ---
  for (int j = 0; j < 8; ++j) {
    const int n = 16 * j + lo;
    const float b  = be1[n];
    const float wd = w1d[n];
    v8f acc;
#pragma unroll
    for (int g = 0; g < 8; ++g) acc[g] = b + d2v[g] * wd;
#pragma unroll
    for (int ks = 0; ks < 4; ++ks) {
      v16bf a0 = frag_a(&Ar[w][lo * D + 32 * ks + 8 * hi]);
      v16bf b0 = frag_b(Wt + 0 * D * D + (size_t)n * D + 32 * ks + 8 * hi);
      acc = wmma_bf16(a0, b0, acc);
      v16bf a1 = frag_a(&Ac[w][lo * D + 32 * ks + 8 * hi]);
      v16bf b1 = frag_b(Wt + 1 * D * D + (size_t)n * D + 32 * ks + 8 * hi);
      acc = wmma_bf16(a1, b1, acc);
    }
#pragma unroll
    for (int g = 0; g < 8; ++g)
      Hid[w][(8 * hi + g) * D + n] = cvt1(fmaxf(acc[g], 0.0f));
  }

  // --- layer 2: messages = Hid@We2 + be2 ; scatter-add into agg ---
  for (int j = 0; j < 8; ++j) {
    const int n = 16 * j + lo;
    const float b = be2[n];
    v8f acc;
#pragma unroll
    for (int g = 0; g < 8; ++g) acc[g] = b;
#pragma unroll
    for (int ks = 0; ks < 4; ++ks) {
      v16bf a = frag_a(&Hid[w][lo * D + 32 * ks + 8 * hi]);
      v16bf bb = frag_b(Wt + 2 * D * D + (size_t)n * D + 32 * ks + 8 * hi);
      acc = wmma_bf16(a, bb, acc);
    }
#pragma unroll
    for (int g = 0; g < 8; ++g) {
      float m = acc[g];
      Msg[(8 * hi + g) * D + n] = cvt1(m);
      atomicAdd(&agg[(size_t)rg[g] * D + n], m);
    }
  }

  // --- coord MLP: cw = relu(msg@Wc1 + bc1) @ Wc2 ---
  float part[8];
#pragma unroll
  for (int g = 0; g < 8; ++g) part[g] = 0.0f;
  for (int j = 0; j < 8; ++j) {
    const int n = 16 * j + lo;
    const float b = bc1[n];
    v8f acc;
#pragma unroll
    for (int g = 0; g < 8; ++g) acc[g] = b;
#pragma unroll
    for (int ks = 0; ks < 4; ++ks) {
      v16bf a = frag_a(&Msg[lo * D + 32 * ks + 8 * hi]);
      v16bf bb = frag_b(Wt + 3 * D * D + (size_t)n * D + 32 * ks + 8 * hi);
      acc = wmma_bf16(a, bb, acc);
    }
    const float wn = Wc2[n];
#pragma unroll
    for (int g = 0; g < 8; ++g) part[g] += fmaxf(acc[g], 0.0f) * wn;
  }
  // reduce over the 16 lanes sharing the same half-wave (hi)
#pragma unroll
  for (int m = 1; m <= 8; m <<= 1) {
#pragma unroll
    for (int g = 0; g < 8; ++g) part[g] += __shfl_xor(part[g], m, 32);
  }
  // one lane per edge row writes the coord update
#pragma unroll
  for (int g = 0; g < 8; ++g) {
    if (lo == g) {
      const int   M  = 8 * hi + g;
      const float cw = part[g];
      const int   r  = rg[g];
      atomicAdd(&out_c[(size_t)r * 3 + 0], cw * Rel[w][M][0]);
      atomicAdd(&out_c[(size_t)r * 3 + 1], cw * Rel[w][M][1]);
      atomicAdd(&out_c[(size_t)r * 3 + 2], cw * Rel[w][M][2]);
    }
  }
}

// ---------------------------------------------------------------------------
// Node kernel: h_new = relu([h, agg]@Wn1 + bn1)@Wn2 + bn2, 16 nodes per wave.
// ---------------------------------------------------------------------------
__global__ __launch_bounds__(TPB) void node_kernel(
    const float* __restrict__ h, const float* __restrict__ agg,
    const float* __restrict__ bn1, const float* __restrict__ bn2,
    const u16* __restrict__ Wt, float* __restrict__ out_h,
    int Nn, int nTiles) {
  __shared__ u16 Ah [WPB][16 * D];
  __shared__ u16 Ag [WPB][16 * D];
  __shared__ u16 Hid[WPB][16 * D];

  const int w  = threadIdx.x >> 5;
  const int L  = threadIdx.x & 31;
  const int lo = L & 15;
  const int hi = L >> 4;
  const int tile = blockIdx.x * WPB + w;
  if (tile >= nTiles) return;

  for (int r = 0; r < 16; ++r) {
    size_t node = (size_t)tile * 16 + r;
    if (node >= (size_t)Nn) node = Nn - 1;    // N%16==0 in practice
    float4 xh = *(const float4*)(h   + node * D + 4 * L);
    float4 xa = *(const float4*)(agg + node * D + 4 * L);
    uint2 ph, pa;
    ph.x = cvt2(xh.x, xh.y);
    ph.y = cvt2(xh.z, xh.w);
    pa.x = cvt2(xa.x, xa.y);
    pa.y = cvt2(xa.z, xa.w);
    *(uint2*)(&Ah[w][r * D + 4 * L]) = ph;
    *(uint2*)(&Ag[w][r * D + 4 * L]) = pa;
  }

  for (int j = 0; j < 8; ++j) {
    const int n = 16 * j + lo;
    const float b = bn1[n];
    v8f acc;
#pragma unroll
    for (int g = 0; g < 8; ++g) acc[g] = b;
#pragma unroll
    for (int ks = 0; ks < 4; ++ks) {
      v16bf a0 = frag_a(&Ah[w][lo * D + 32 * ks + 8 * hi]);
      v16bf b0 = frag_b(Wt + 4 * D * D + (size_t)n * D + 32 * ks + 8 * hi);
      acc = wmma_bf16(a0, b0, acc);
      v16bf a1 = frag_a(&Ag[w][lo * D + 32 * ks + 8 * hi]);
      v16bf b1 = frag_b(Wt + 5 * D * D + (size_t)n * D + 32 * ks + 8 * hi);
      acc = wmma_bf16(a1, b1, acc);
    }
#pragma unroll
    for (int g = 0; g < 8; ++g)
      Hid[w][(8 * hi + g) * D + n] = cvt1(fmaxf(acc[g], 0.0f));
  }

  for (int j = 0; j < 8; ++j) {
    const int n = 16 * j + lo;
    const float b = bn2[n];
    v8f acc;
#pragma unroll
    for (int g = 0; g < 8; ++g) acc[g] = b;
#pragma unroll
    for (int ks = 0; ks < 4; ++ks) {
      v16bf a = frag_a(&Hid[w][lo * D + 32 * ks + 8 * hi]);
      v16bf bb = frag_b(Wt + 6 * D * D + (size_t)n * D + 32 * ks + 8 * hi);
      acc = wmma_bf16(a, bb, acc);
    }
#pragma unroll
    for (int g = 0; g < 8; ++g) {
      int row = tile * 16 + 8 * hi + g;
      if (row < Nn) out_h[(size_t)row * D + n] = acc[g];
    }
  }
}

// ---------------------------------------------------------------------------
extern "C" void kernel_launch(void* const* d_in, const int* in_sizes, int n_in,
                              void* d_out, int out_size, void* d_ws, size_t ws_size,
                              hipStream_t stream) {
  (void)n_in; (void)out_size; (void)ws_size;
  const float* h      = (const float*)d_in[0];
  const float* coords = (const float*)d_in[1];
  const int*   eidx   = (const int*)  d_in[2];
  const float* We1    = (const float*)d_in[3];
  const float* be1    = (const float*)d_in[4];
  const float* We2    = (const float*)d_in[5];
  const float* be2    = (const float*)d_in[6];
  const float* Wn1    = (const float*)d_in[7];
  const float* bn1    = (const float*)d_in[8];
  const float* Wn2    = (const float*)d_in[9];
  const float* bn2    = (const float*)d_in[10];
  const float* Wc1    = (const float*)d_in[11];
  const float* bc1    = (const float*)d_in[12];
  const float* Wc2    = (const float*)d_in[13];

  const int N = in_sizes[0] / D;     // 50000
  const int E = in_sizes[2] / 2;     // 600000

  float* agg = (float*)d_ws;                                        // [N, D] f32
  u16*   Wt  = (u16*)((char*)d_ws + (size_t)N * D * sizeof(float)); // 7x[D,D] bf16

  float* out_h = (float*)d_out;                 // [N, D]
  float* out_c = out_h + (size_t)N * D;         // [N, 3]

  prep_weights<<<(7 * D * D + 255) / 256, 256, 0, stream>>>(We1, We2, Wc1, Wn1, Wn2, Wt);
  init_out<<<(N * D + 255) / 256, 256, 0, stream>>>(coords, agg, out_c, N);

  const int eTiles = (E + 15) / 16;
  edge_kernel<<<(eTiles + WPB - 1) / WPB, TPB, 0, stream>>>(
      h, coords, eidx, We1, be1, be2, bc1, Wc2, Wt, agg, out_c, E, eTiles);

  const int nTiles = (N + 15) / 16;
  node_kernel<<<(nTiles + WPB - 1) / WPB, TPB, 0, stream>>>(
      h, agg, bn1, bn2, Wt, out_h, N, nTiles);
}
